// GNN_86878598464102
// MI455X (gfx1250) — matrix operand order repaired and checked by reading, not compiled, not run on hardware
//
#include <hip/hip_runtime.h>

#define MAX_NN 21
#define BATCH  1024
#define NODE_ROWS (BATCH * MAX_NN)            // 21504
#define EDGE_ROWS (BATCH * MAX_NN * MAX_NN)   // 451584

typedef __attribute__((ext_vector_type(8)))  __bf16 v8bf;
typedef __attribute__((ext_vector_type(16))) __bf16 v16bf;
typedef __attribute__((ext_vector_type(8)))  float  v8f;

// ---------------------------------------------------------------------------
// WMMA helpers (CDNA5 wave32, V_WMMA_F32_16X16X32_BF16)
// ---------------------------------------------------------------------------
__device__ __forceinline__ v8f wmma_bf16(v16bf a, v16bf b, v8f c) {
  return __builtin_amdgcn_wmma_f32_16x16x32_bf16(false, a, false, b, (short)0, c,
                                                 false, false);
}

// A fragment: 16x32 bf16 tile from row-major source (leading dim ld, elems).
// ISA layout: lane m = l&15; lane-half h=l>>4: elems 0..7 -> K = h*8+{0..7},
// elems 8..15 -> K = 16 + h*8 + {0..7}.
__device__ __forceinline__ v16bf load_a_16x32(const __bf16* A, int ld, int m0, int kb) {
  const int lane = (int)(threadIdx.x & 31u);
  const int m = m0 + (lane & 15);
  const int h = lane >> 4;
  const __bf16* row = A + (size_t)m * ld + kb;
  v8bf lo = *(const v8bf*)(row + h * 8);
  v8bf hi = *(const v8bf*)(row + 16 + h * 8);
  v16bf a;
#pragma unroll
  for (int t = 0; t < 8; ++t) { a[t] = lo[t]; a[8 + t] = hi[t]; }
  return a;
}

// B fragment: 32x16 (KxN). Weights are torch layout [N][K] (ldk=K), so
// B[k][n] = W[n][k]; lane n = l&15, lane-half h: elems 0..7 -> K = h*8+{0..7},
// elems 8..15 -> K = 16 + h*8 + {0..7}.
__device__ __forceinline__ v16bf load_b_32x16(const __bf16* W, int ldk, int n0, int kb) {
  const int lane = (int)(threadIdx.x & 31u);
  const int n = n0 + (lane & 15);
  const int h = lane >> 4;
  const __bf16* q = W + (size_t)n * ldk + kb + h * 8;
  v8bf lo = *(const v8bf*)q;
  v8bf hi = *(const v8bf*)(q + 16);
  v16bf b;
#pragma unroll
  for (int t = 0; t < 8; ++t) { b[t] = lo[t]; b[8 + t] = hi[t]; }
  return b;
}

template <int NT>
__device__ __forceinline__ void zero_acc(v8f (&acc)[NT]) {
#pragma unroll
  for (int t = 0; t < NT; ++t)
#pragma unroll
    for (int r = 0; r < 8; ++r) acc[t][r] = 0.f;
}

// One wave computes a 16-row x (NT*16)-col strip: A from LDS, B from global bf16.
template <int KD, int NT>
__device__ __forceinline__ void wave_gemm(const __bf16* A, int lda, int m0,
                                          const __bf16* W, int n0, v8f (&acc)[NT]) {
#pragma unroll
  for (int kb = 0; kb < KD; kb += 32) {
    v16bf a = load_a_16x32(A, lda, m0, kb);
#pragma unroll
    for (int t = 0; t < NT; ++t) {
      v16bf b = load_b_32x16(W, KD, n0 + t * 16, kb);
      acc[t] = wmma_bf16(a, b, acc[t]);
    }
  }
}

// D epilogue -> LDS bf16 with bias + ReLU. C/D layout: lane n=l&15, VGPR r -> M=r+8*(l>>4).
template <int NT>
__device__ __forceinline__ void store_bias_relu(const v8f (&acc)[NT], const float* bias,
                                                __bf16* dst, int ld, int m0, int n0) {
  const int lane = (int)(threadIdx.x & 31u);
  const int h = lane >> 4, nl = lane & 15;
#pragma unroll
  for (int t = 0; t < NT; ++t) {
    const int n = n0 + t * 16 + nl;
    const float b = bias[n];
#pragma unroll
    for (int r = 0; r < 8; ++r) {
      const int m = m0 + r + 8 * h;
      float v = acc[t][r] + b;
      dst[m * ld + n] = (__bf16)(v > 0.f ? v : 0.f);
    }
  }
}

// ---------------------------------------------------------------------------
// K0: fp32 -> bf16 weight conversion
// ---------------------------------------------------------------------------
__global__ void cvt_bf16_kernel(const float* __restrict__ src, __bf16* __restrict__ dst,
                                int n) {
  int i = (int)(blockIdx.x * 256u + threadIdx.x);
  if (i < n) dst[i] = (__bf16)src[i];
}

// ---------------------------------------------------------------------------
// K1: node path. Per 64-row tile: MLP(64->64->64->128)+mask -> node (fp32 out),
// then f_i = node@Wi.T, f_j = node@Wj.T (bf16 out). node is loop-invariant in
// the reference, so f_i/f_j are computed exactly once here.
// ---------------------------------------------------------------------------
__global__ __launch_bounds__(256) void node_path_kernel(
    const float* __restrict__ nf, const float* __restrict__ maskV,
    const __bf16* __restrict__ W0, const float* __restrict__ b0,
    const __bf16* __restrict__ W1, const float* __restrict__ b1,
    const __bf16* __restrict__ W2, const float* __restrict__ b2,
    const __bf16* __restrict__ Wi, const __bf16* __restrict__ Wj,
    float* __restrict__ node_out, __bf16* __restrict__ fi, __bf16* __restrict__ fj) {
  __shared__ __attribute__((aligned(16))) __bf16 bufA[64 * 72];
  __shared__ __attribute__((aligned(16))) __bf16 bufB[64 * 72];
  __shared__ __attribute__((aligned(16))) __bf16 bufC[64 * 136];
  __shared__ float mV[64];

  const int tid = (int)threadIdx.x;
  const int row0 = (int)blockIdx.x * 64;
  const int wave = tid >> 5, lane = tid & 31;
  const int m0 = (wave >> 1) * 16;
  const int h = lane >> 4, nl = lane & 15;
  const int n0w = (wave & 1) * 64;

  {  // stage 64x64 fp32 -> bf16 LDS (ld 72)
    const int m = tid >> 2, c0 = (tid & 3) * 16;
    const float4* s4 = (const float4*)(nf + (size_t)(row0 + m) * 64 + c0);
#pragma unroll
    for (int q = 0; q < 4; ++q) {
      float4 v = s4[q];
      bufA[m * 72 + c0 + q * 4 + 0] = (__bf16)v.x;
      bufA[m * 72 + c0 + q * 4 + 1] = (__bf16)v.y;
      bufA[m * 72 + c0 + q * 4 + 2] = (__bf16)v.z;
      bufA[m * 72 + c0 + q * 4 + 3] = (__bf16)v.w;
    }
    if (tid < 64) mV[tid] = maskV[row0 + tid];
  }
  __syncthreads();

  {  // L0: 64->64, ReLU
    v8f acc[2]; zero_acc(acc);
    const int n0 = (wave & 1) * 32;
    wave_gemm<64, 2>(bufA, 72, m0, W0, n0, acc);
    store_bias_relu(acc, b0, bufB, 72, m0, n0);
  }
  __syncthreads();
  {  // L1: 64->64, ReLU
    v8f acc[2]; zero_acc(acc);
    const int n0 = (wave & 1) * 32;
    wave_gemm<64, 2>(bufB, 72, m0, W1, n0, acc);
    store_bias_relu(acc, b1, bufA, 72, m0, n0);
  }
  __syncthreads();
  {  // L2: 64->128, bias (no relu), mask_V; fp32 to d_out + bf16 to bufC
    v8f acc[4]; zero_acc(acc);
    wave_gemm<64, 4>(bufA, 72, m0, W2, n0w, acc);
#pragma unroll
    for (int t = 0; t < 4; ++t) {
      const int n = n0w + t * 16 + nl;
      const float b = b2[n];
#pragma unroll
      for (int r = 0; r < 8; ++r) {
        const int m = m0 + r + 8 * h;
        const float v = (acc[t][r] + b) * mV[m];
        node_out[(size_t)(row0 + m) * 128 + n] = v;
        bufC[m * 136 + n] = (__bf16)v;
      }
    }
  }
  __syncthreads();
  {  // f_i = node @ Wi.T
    v8f acc[4]; zero_acc(acc);
    wave_gemm<128, 4>(bufC, 136, m0, Wi, n0w, acc);
#pragma unroll
    for (int t = 0; t < 4; ++t) {
      const int n = n0w + t * 16 + nl;
#pragma unroll
      for (int r = 0; r < 8; ++r)
        fi[(size_t)(row0 + m0 + r + 8 * h) * 128 + n] = (__bf16)acc[t][r];
    }
  }
  {  // f_j = node @ Wj.T
    v8f acc[4]; zero_acc(acc);
    wave_gemm<128, 4>(bufC, 136, m0, Wj, n0w, acc);
#pragma unroll
    for (int t = 0; t < 4; ++t) {
      const int n = n0w + t * 16 + nl;
#pragma unroll
      for (int r = 0; r < 8; ++r)
        fj[(size_t)(row0 + m0 + r + 8 * h) * 128 + n] = (__bf16)acc[t][r];
    }
  }
}

// ---------------------------------------------------------------------------
// K2: fully fused edge path. Per 64-row tile: edge input MLP (32->32->32->128,
// mask_E) then GNN_DEPTH=3 iterations of
//   edge += mask_E * MLP_de(relu(edge@Wij.T + bij + f_i[b,j] + f_j[b,i]))
// entirely in LDS/registers; `edge` value lives in registers across all
// iterations, one fp32 store at the end.
// ---------------------------------------------------------------------------
__global__ __launch_bounds__(256) void gnn_edge_kernel(
    const float* __restrict__ ea, const float* __restrict__ maskE,
    const __bf16* __restrict__ eW0, const float* __restrict__ eb0,
    const __bf16* __restrict__ eW1, const float* __restrict__ eb1,
    const __bf16* __restrict__ eW2, const float* __restrict__ eb2,
    const __bf16* __restrict__ fi, const __bf16* __restrict__ fj,
    const __bf16* __restrict__ Wij, const float* __restrict__ bij,
    const __bf16* __restrict__ dW0, const float* __restrict__ db0,
    const __bf16* __restrict__ dW1, const float* __restrict__ db1,
    const __bf16* __restrict__ dW2, const float* __restrict__ db2,
    float* __restrict__ edge_out) {
  __shared__ __attribute__((aligned(16))) __bf16 bufA[64 * 136];
  __shared__ __attribute__((aligned(16))) __bf16 bufB[64 * 136];
  __shared__ __attribute__((aligned(16))) __bf16 fsum[64 * 136];
  __shared__ float mE[64];

  const int tid = (int)threadIdx.x;
  const int row0 = (int)blockIdx.x * 64;
  const int wave = tid >> 5, lane = tid & 31;
  const int m0 = (wave >> 1) * 16;
  const int h = lane >> 4, nl = lane & 15;
  const int n0w = (wave & 1) * 64;

  {  // stage edge_attr 64x32 -> bf16 (ld 40); build fsum (loop-invariant); mE
    const int m = tid >> 2;
    const int c0 = (tid & 3) * 8;
    const float4* s4 = (const float4*)(ea + (size_t)(row0 + m) * 32 + c0);
    float4 v0 = s4[0], v1 = s4[1];
    bufA[m * 40 + c0 + 0] = (__bf16)v0.x;
    bufA[m * 40 + c0 + 1] = (__bf16)v0.y;
    bufA[m * 40 + c0 + 2] = (__bf16)v0.z;
    bufA[m * 40 + c0 + 3] = (__bf16)v0.w;
    bufA[m * 40 + c0 + 4] = (__bf16)v1.x;
    bufA[m * 40 + c0 + 5] = (__bf16)v1.y;
    bufA[m * 40 + c0 + 6] = (__bf16)v1.z;
    bufA[m * 40 + c0 + 7] = (__bf16)v1.w;

    const int f0 = (tid & 3) * 32;
    const int e = row0 + m;
    const int b = e / (MAX_NN * MAX_NN);
    const int rem = e - b * (MAX_NN * MAX_NN);
    const int i = rem / MAX_NN;
    const int j = rem - i * MAX_NN;
    const __bf16* fip = fi + (size_t)(b * MAX_NN + j) * 128 + f0;
    const __bf16* fjp = fj + (size_t)(b * MAX_NN + i) * 128 + f0;
#pragma unroll
    for (int q = 0; q < 4; ++q) {
      v8bf x = *(const v8bf*)(fip + q * 8);
      v8bf y = *(const v8bf*)(fjp + q * 8);
#pragma unroll
      for (int t = 0; t < 8; ++t)
        fsum[m * 136 + f0 + q * 8 + t] =
            (__bf16)(bij[f0 + q * 8 + t] + (float)x[t] + (float)y[t]);
    }
    if (tid < 64) mE[tid] = maskE[row0 + tid];
  }
  __syncthreads();

  // --- edge input MLP ---
  {  // eL0: 32->32, ReLU
    v8f acc[1]; zero_acc(acc);
    const int n0 = (wave & 1) * 16;
    wave_gemm<32, 1>(bufA, 40, m0, eW0, n0, acc);
    store_bias_relu(acc, eb0, bufB, 40, m0, n0);
  }
  __syncthreads();
  {  // eL1: 32->32, ReLU
    v8f acc[1]; zero_acc(acc);
    const int n0 = (wave & 1) * 16;
    wave_gemm<32, 1>(bufB, 40, m0, eW1, n0, acc);
    store_bias_relu(acc, eb1, bufA, 40, m0, n0);
  }
  __syncthreads();

  float eold[4][8];  // running edge value in registers (epilogue layout)
  {  // eL2: 32->128, bias, mask_E
    v8f acc[4]; zero_acc(acc);
    wave_gemm<32, 4>(bufA, 40, m0, eW2, n0w, acc);
    __syncthreads();  // all ld40 reads of bufA done before ld136 relayout write
#pragma unroll
    for (int t = 0; t < 4; ++t) {
      const int n = n0w + t * 16 + nl;
      const float b = eb2[n];
#pragma unroll
      for (int r = 0; r < 8; ++r) {
        const int m = m0 + r + 8 * h;
        const float v = (acc[t][r] + b) * mE[m];
        eold[t][r] = v;
        bufA[m * 136 + n] = (__bf16)v;
      }
    }
  }
  __syncthreads();

  // --- GNN iterations (independent per row; no global sync needed) ---
#pragma unroll 1
  for (int it = 0; it < 3; ++it) {
    {  // s = relu(edge@Wij.T + fsum)
      v8f acc[4]; zero_acc(acc);
      wave_gemm<128, 4>(bufA, 136, m0, Wij, n0w, acc);
#pragma unroll
      for (int t = 0; t < 4; ++t) {
        const int n = n0w + t * 16 + nl;
#pragma unroll
        for (int r = 0; r < 8; ++r) {
          const int m = m0 + r + 8 * h;
          float v = acc[t][r] + (float)fsum[m * 136 + n];
          bufB[m * 136 + n] = (__bf16)(v > 0.f ? v : 0.f);
        }
      }
    }
    __syncthreads();
    {  // h1 = relu(s@dW0.T + db0)
      v8f acc[4]; zero_acc(acc);
      wave_gemm<128, 4>(bufB, 136, m0, dW0, n0w, acc);
      store_bias_relu(acc, db0, bufA, 136, m0, n0w);
    }
    __syncthreads();
    {  // h2 = relu(h1@dW1.T + db1)
      v8f acc[4]; zero_acc(acc);
      wave_gemm<128, 4>(bufA, 136, m0, dW1, n0w, acc);
      store_bias_relu(acc, db1, bufB, 136, m0, n0w);
    }
    __syncthreads();
    {  // dE = h2@dW2.T + db2 ; edge += mask_E * dE
      v8f acc[4]; zero_acc(acc);
      wave_gemm<128, 4>(bufB, 136, m0, dW2, n0w, acc);
#pragma unroll
      for (int t = 0; t < 4; ++t) {
        const int n = n0w + t * 16 + nl;
        const float b = db2[n];
#pragma unroll
        for (int r = 0; r < 8; ++r) {
          const int m = m0 + r + 8 * h;
          const float v = eold[t][r] + mE[m] * (acc[t][r] + b);
          eold[t][r] = v;
          bufA[m * 136 + n] = (__bf16)v;
        }
      }
    }
    __syncthreads();
  }

  // final fp32 store of edge (coalesced 64B per half-wave)
#pragma unroll
  for (int t = 0; t < 4; ++t) {
    const int n = n0w + t * 16 + nl;
#pragma unroll
    for (int r = 0; r < 8; ++r) {
      const int m = m0 + r + 8 * h;
      edge_out[(size_t)(row0 + m) * 128 + n] = eold[t][r];
    }
  }
}

// ---------------------------------------------------------------------------
// Host-side launcher
// ---------------------------------------------------------------------------
extern "C" void kernel_launch(void* const* d_in, const int* in_sizes, int n_in,
                              void* d_out, int out_size, void* d_ws, size_t ws_size,
                              hipStream_t stream) {
  (void)in_sizes; (void)n_in; (void)out_size; (void)ws_size;

  // Input order (setup_inputs dict, lists expanded in order):
  const float* node_feats = (const float*)d_in[0];
  const float* edge_attr  = (const float*)d_in[1];
  const float* mask_V     = (const float*)d_in[2];
  const float* mask_E     = (const float*)d_in[3];
  // 4..6 node_W[0..2], 7..9 node_b[0..2], 10..12 edge_W[0..2], 13..15 edge_b[0..2]
  const float* node_b0 = (const float*)d_in[7];
  const float* node_b1 = (const float*)d_in[8];
  const float* node_b2 = (const float*)d_in[9];
  const float* edge_b0 = (const float*)d_in[13];
  const float* edge_b1 = (const float*)d_in[14];
  const float* edge_b2 = (const float*)d_in[15];
  const float* pf_bij  = (const float*)d_in[17];
  const float* de_b0   = (const float*)d_in[23];
  const float* de_b1   = (const float*)d_in[24];
  const float* de_b2   = (const float*)d_in[25];

  // workspace layout (bf16 weights + f_i / f_j)
  char* ws = (char*)d_ws;
  size_t off = 0;
  auto walloc = [&](size_t elems) -> __bf16* {
    __bf16* p = (__bf16*)(ws + off);
    off = (off + elems * sizeof(__bf16) + 255) & ~(size_t)255;
    return p;
  };
  __bf16* nW0 = walloc(64 * 64);
  __bf16* nW1 = walloc(64 * 64);
  __bf16* nW2 = walloc(128 * 64);
  __bf16* eW0 = walloc(32 * 32);
  __bf16* eW1 = walloc(32 * 32);
  __bf16* eW2 = walloc(128 * 32);
  __bf16* Wij = walloc(128 * 128);
  __bf16* Wi  = walloc(128 * 128);
  __bf16* Wj  = walloc(128 * 128);
  __bf16* dW0 = walloc(128 * 128);
  __bf16* dW1 = walloc(128 * 128);
  __bf16* dW2 = walloc(128 * 128);
  __bf16* fi  = walloc((size_t)NODE_ROWS * 128);
  __bf16* fj  = walloc((size_t)NODE_ROWS * 128);

  auto cvt = [&](int idx, size_t n, __bf16* dst) {
    cvt_bf16_kernel<<<(unsigned)((n + 255) / 256), 256, 0, stream>>>(
        (const float*)d_in[idx], dst, (int)n);
  };
  cvt(4, 64 * 64, nW0);   cvt(5, 64 * 64, nW1);   cvt(6, 128 * 64, nW2);
  cvt(10, 32 * 32, eW0);  cvt(11, 32 * 32, eW1);  cvt(12, 128 * 32, eW2);
  cvt(16, 128 * 128, Wij);
  cvt(18, 128 * 128, Wi); cvt(19, 128 * 128, Wj);
  cvt(20, 128 * 128, dW0); cvt(21, 128 * 128, dW1); cvt(22, 128 * 128, dW2);

  float* node_out = (float*)d_out;                          // [21504,128]
  float* edge_out = (float*)d_out + (size_t)NODE_ROWS * 128;  // [451584,128]

  node_path_kernel<<<NODE_ROWS / 64, 256, 0, stream>>>(
      node_feats, mask_V, nW0, node_b0, nW1, node_b1, nW2, node_b2, Wi, Wj,
      node_out, fi, fj);

  gnn_edge_kernel<<<EDGE_ROWS / 64, 256, 0, stream>>>(
      edge_attr, mask_E, eW0, edge_b0, eW1, edge_b1, eW2, edge_b2, fi, fj,
      Wij, pf_bij, dW0, de_b0, dW1, de_b1, dW2, de_b2, edge_out);
}
